// BaseConchGS_16406775071376
// MI455X (gfx1250) — matrix operand order repaired and checked by reading, not compiled, not run on hardware
//
#include <hip/hip_runtime.h>
#include <hip/hip_bf16.h>

// ---------------------------------------------------------------------------
// Problem constants (match reference)
// ---------------------------------------------------------------------------
#define BSEED   4096
#define SNEI    32
#define DDIM    128
#define DEDGE   64
#define NMP     2
#define NEDGEROW (BSEED * SNEI)     // 131072 gathered edge rows

typedef __bf16 bf16_t;
typedef __attribute__((ext_vector_type(16))) __bf16 v16bf;
typedef __attribute__((ext_vector_type(8)))  __bf16 bf16x8;
typedef __attribute__((ext_vector_type(8)))  float  v8f;

// ---------------------------------------------------------------------------
// Small helper kernels: gathers, means, weight folding (all tiny cost)
// ---------------------------------------------------------------------------

// F0[b*128+t] = bf16(feats[ids[b]][t])
__global__ __launch_bounds__(256)
void k_gather_f0(const int* __restrict__ ids, const float* __restrict__ feats,
                 bf16_t* __restrict__ F0) {
    int idx = blockIdx.x * 256 + threadIdx.x;      // B*128 total
    int b = idx >> 7, t = idx & 127;
    F0[idx] = (bf16_t)feats[(long)ids[b] * DDIM + t];
}

// Per seed: gather 32 raw edge-embedding rows (64 f32) -> Egraw bf16,
// and their mean -> Em bf16. One block (64 threads) per seed.
__global__ __launch_bounds__(64)
void k_gather_edges(const int* __restrict__ ids, const int* __restrict__ n2e,
                    const float* __restrict__ eemb,
                    bf16_t* __restrict__ Egraw, bf16_t* __restrict__ Em) {
    int b = blockIdx.x, t = threadIdx.x;           // t in [0,64)
    int id = ids[b];
    const int* row = n2e + (long)id * SNEI;
    float acc = 0.f;
    for (int s = 0; s < SNEI; ++s) {
        int eid = row[s];
        float v = eemb[(long)eid * DEDGE + t];
        Egraw[((long)(b * SNEI + s)) * DEDGE + t] = (bf16_t)v;
        acc += v;
    }
    Em[(long)b * DEDGE + t] = (bf16_t)(acc * (1.f / 32.f));
}

// Per gathered edge row: mean of the two endpoint feature rows (raw, pre-prep).
__global__ __launch_bounds__(128)
void k_gather_mean_nodes(const int* __restrict__ ids, const int* __restrict__ n2e,
                         const int* __restrict__ adj, const float* __restrict__ feats,
                         bf16_t* __restrict__ Fmean) {
    int rowi = blockIdx.x;                          // 0 .. NEDGEROW-1
    int b = rowi >> 5, s = rowi & 31, t = threadIdx.x;
    int id  = ids[b];
    int eid = n2e[(long)id * SNEI + s];
    int n0  = adj[(long)eid * 2 + 0];
    int n1  = adj[(long)eid * 2 + 1];
    float v = 0.5f * (feats[(long)n0 * DDIM + t] + feats[(long)n1 * DDIM + t]);
    Fmean[(long)rowi * DDIM + t] = (bf16_t)v;
}

// m1[b] = mean over 32 fan-out rows of nxt1 (post-relu, cannot be folded).
__global__ __launch_bounds__(128)
void k_mean32(const bf16_t* __restrict__ nxt1, bf16_t* __restrict__ m1) {
    int b = blockIdx.x, t = threadIdx.x;
    float acc = 0.f;
    for (int s = 0; s < SNEI; ++s)
        acc += (float)nxt1[((long)(b * SNEI + s)) * DDIM + t];
    m1[(long)b * DDIM + t] = (bf16_t)(acc * (1.f / 32.f));
}

// C[KA,128] = bf16( A[KA,128] @ W[128,128] )   (weight folding, f32 accumulate)
__global__ __launch_bounds__(256)
void k_fold_w(const float* __restrict__ A, const float* __restrict__ W,
              bf16_t* __restrict__ C, int KA) {
    int idx = blockIdx.x * 256 + threadIdx.x;
    if (idx >= KA * DDIM) return;
    int i = idx >> 7, j = idx & 127;
    float s = 0.f;
    for (int k = 0; k < DDIM; ++k) s += A[i * DDIM + k] * W[k * DDIM + j];
    C[idx] = (bf16_t)s;
}

__global__ __launch_bounds__(256)
void k_cvt_bf16(const float* __restrict__ in, bf16_t* __restrict__ out, int n) {
    int i = blockIdx.x * 256 + threadIdx.x;
    if (i < n) out[i] = (bf16_t)in[i];
}

// ---------------------------------------------------------------------------
// Fused dual-source WMMA GEMM (templated on K1,K2 so both K loops fully
// unroll with immediate offsets; no per-iteration operand select):
//   C[M,128] = relu( A1[M,K1] @ B1[K1,128] + A2[M,K2] @ B2[K2,128] )
// Block = 128 threads (4 waves), 64 rows per block, 16x128 strip per wave.
// B1|B2 staged in LDS pre-swizzled into v_wmma_f32_16x16x32_bf16 B-fragment
// layout (lanes 0-15: K=0..7,16..23, lanes 16-31: K=8..15,24..31, N = lane&15; element e
// of the per-lane v16bf holds K = (e>>3)*16 + half*8 + (e&7)). A fragment loaded straight
// from row-major bf16: element e -> K = (e>>3)*16 + half*8 + (e&7), i.e. two
// contiguous 16-byte chunks per lane per K-step (ISA 16-bit A 16x32 table).
// ---------------------------------------------------------------------------
template<int K1, int K2>
__global__ __launch_bounds__(128)
void k_dual_gemm_wmma(const bf16_t* __restrict__ A1, const bf16_t* __restrict__ B1,
                      const bf16_t* __restrict__ A2, const bf16_t* __restrict__ B2,
                      bf16_t* __restrict__ outBf,      // [M,128] or null
                      float* __restrict__ outF32,      // row stride ldc, or null
                      int ldc, int colBase) {
    constexpr int NKT1 = K1 >> 5;
    constexpr int NKT2 = K2 >> 5;
    constexpr int NKT  = NKT1 + NKT2;

    __shared__ bf16_t Bfrag[NKT * 8 * 32 * 16];

    // ---- stage B1|B2 into LDS in fragment order -------------------------
    constexpr int TOTAL = (K1 + K2) * DDIM;
    for (int idx = threadIdx.x; idx < TOTAL; idx += 128) {
        int k, n, ktbase;
        const bf16_t* src;
        if (idx < K1 * DDIM) { k = idx >> 7; n = idx & 127; src = B1; ktbase = 0; }
        else { int r = idx - K1 * DDIM; k = r >> 7; n = r & 127; src = B2; ktbase = NKT1; }
        bf16_t v = src[(long)k * DDIM + n];
        int kt = ktbase + (k >> 5);
        int kk = k & 31;
        int h  = (kk >> 3) & 1;    // which lane-half holds this K
        int e  = (kk & 7) | ((kk >> 4) << 3);   // element within the lane's v16bf
        int nt = n >> 4;           // N tile (8 tiles of 16)
        int c  = n & 15;           // column within tile == lane&15
        Bfrag[(((kt * 8 + nt) * 32) + (h * 16 + c)) * 16 + e] = v;
    }
    __syncthreads();

    // ---- per-wave 16x128 output strip -----------------------------------
    const int wave = threadIdx.x >> 5;
    const int lane = threadIdx.x & 31;
    const int half = lane >> 4;
    const int r    = lane & 15;
    const long rowBase = (long)blockIdx.x * 64 + wave * 16;

    const bf16_t* __restrict__ A1row = A1 + (rowBase + r) * (long)K1;
    const bf16_t* __restrict__ A2row = A2 + (rowBase + r) * (long)K2;
    const bf16_t* __restrict__ myB   = &Bfrag[lane * 16];

    v8f acc[8];
    #pragma unroll
    for (int nt = 0; nt < 8; ++nt)
        #pragma unroll
        for (int i = 0; i < 8; ++i) acc[nt][i] = 0.f;

    #pragma unroll
    for (int kt = 0; kt < NKT; ++kt) {
        const bf16_t* Ap = (kt < NKT1) ? (A1row + kt * 32)
                                       : (A2row + (kt - NKT1) * 32);
        bf16x8 alo = *(const bf16x8*)(Ap + half * 8);
        bf16x8 ahi = *(const bf16x8*)(Ap + 16 + half * 8);
        v16bf a;
        #pragma unroll
        for (int e = 0; e < 8; ++e) { a[e] = alo[e]; a[e + 8] = ahi[e]; }

        #pragma unroll
        for (int nt = 0; nt < 8; ++nt) {
            const v16bf b = *(const v16bf*)(myB + (kt * 8 + nt) * 32 * 16);
            acc[nt] = __builtin_amdgcn_wmma_f32_16x16x32_bf16(
                          false, a, false, b, (short)0, acc[nt], false, false);
        }
    }

    // ---- relu epilogue: C VGPR i -> row = base + i + half*8, col = nt*16+r
    #pragma unroll
    for (int nt = 0; nt < 8; ++nt) {
        #pragma unroll
        for (int i = 0; i < 8; ++i) {
            long row = rowBase + i + half * 8;
            int  col = nt * 16 + r;
            float v = acc[nt][i];
            v = v > 0.f ? v : 0.f;
            if (outBf)  outBf[row * DDIM + col] = (bf16_t)v;
            if (outF32) outF32[row * (long)ldc + colBase + col] = v;
        }
    }
}

// ---------------------------------------------------------------------------
// Host-side orchestration
// ---------------------------------------------------------------------------
extern "C" void kernel_launch(void* const* d_in, const int* in_sizes, int n_in,
                              void* d_out, int out_size, void* d_ws, size_t ws_size,
                              hipStream_t stream) {
    const int*   ids    = (const int*)d_in[0];
    const float* feats  = (const float*)d_in[1];
    const int*   n2e[2] = {(const int*)d_in[2], (const int*)d_in[3]};
    const int*   adj[2] = {(const int*)d_in[4], (const int*)d_in[5]};
    const float* eemb[2]= {(const float*)d_in[6], (const float*)d_in[7]};
    const float* prepW  = (const float*)d_in[8];
    const float* eprepW = (const float*)d_in[9];
    const float* WnS    = (const float*)d_in[10];
    const float* WnN    = (const float*)d_in[11];
    const float* WeS    = (const float*)d_in[12];
    const float* WeN    = (const float*)d_in[13];
    float* out = (float*)d_out;

    // ---- workspace layout (bf16, 256B aligned) --------------------------
    char* ws = (char*)d_ws;
    size_t off = 0;
    auto alloc = [&](size_t elems) -> bf16_t* {
        bf16_t* p = (bf16_t*)(ws + off);
        off += (elems * sizeof(bf16_t) + 255) & ~(size_t)255;
        return p;
    };
    bf16_t* CnS0 = alloc(128 * 128);   // prep_W @ Wn_self[mp,0]
    bf16_t* CnN0 = alloc( 64 * 128);   // edge_prep_W @ Wn_neigh[mp,0]
    bf16_t* CeS0 = alloc( 64 * 128);   // edge_prep_W @ We_self[mp,0]
    bf16_t* CeN0 = alloc(128 * 128);   // prep_W @ We_neigh[mp,0]
    bf16_t* Wn1s = alloc(128 * 128);   // Wn_self[mp,1] (bf16)
    bf16_t* Wn1n = alloc(128 * 128);   // Wn_neigh[mp,1] (bf16)
    bf16_t* F0   = alloc((size_t)BSEED * DDIM);
    bf16_t* Em   = alloc((size_t)BSEED * DEDGE);
    bf16_t* Eg   = alloc((size_t)NEDGEROW * DEDGE);
    bf16_t* Fm   = alloc((size_t)NEDGEROW * DDIM);
    bf16_t* nxt0 = alloc((size_t)BSEED * DDIM);
    bf16_t* nxt1 = alloc((size_t)NEDGEROW * DDIM);
    bf16_t* m1   = alloc((size_t)BSEED * DDIM);
    (void)ws_size; (void)in_sizes; (void)n_in; (void)out_size;

    // Seed features (metapath independent)
    k_gather_f0<<<(BSEED * DDIM) / 256, 256, 0, stream>>>(ids, feats, F0);

    for (int mp = 0; mp < NMP; ++mp) {
        const float* Wn0s = WnS + ((size_t)(mp * 2 + 0)) * 128 * 128;
        const float* Wn0n = WnN + ((size_t)(mp * 2 + 0)) * 128 * 128;
        const float* We0s = WeS + ((size_t)(mp * 2 + 0)) * 128 * 128;
        const float* We0n = WeN + ((size_t)(mp * 2 + 0)) * 128 * 128;
        const float* Wn1sf= WnS + ((size_t)(mp * 2 + 1)) * 128 * 128;
        const float* Wn1nf= WnN + ((size_t)(mp * 2 + 1)) * 128 * 128;
        const float* epW  = eprepW + (size_t)mp * DEDGE * DDIM;

        // Fold linear prep maps into layer-0 weights (mean commutes with prep)
        k_fold_w<<<(128 * 128 + 255) / 256, 256, 0, stream>>>(prepW, Wn0s, CnS0, 128);
        k_fold_w<<<( 64 * 128 + 255) / 256, 256, 0, stream>>>(epW,   Wn0n, CnN0,  64);
        k_fold_w<<<( 64 * 128 + 255) / 256, 256, 0, stream>>>(epW,   We0s, CeS0,  64);
        k_fold_w<<<(128 * 128 + 255) / 256, 256, 0, stream>>>(prepW, We0n, CeN0, 128);
        k_cvt_bf16<<<(128 * 128 + 255) / 256, 256, 0, stream>>>(Wn1sf, Wn1s, 128 * 128);
        k_cvt_bf16<<<(128 * 128 + 255) / 256, 256, 0, stream>>>(Wn1nf, Wn1n, 128 * 128);

        // Gathers + fan-out means on raw (pre-prep) data
        k_gather_edges<<<BSEED, 64, 0, stream>>>(ids, n2e[mp], eemb[mp], Eg, Em);
        k_gather_mean_nodes<<<NEDGEROW, 128, 0, stream>>>(ids, n2e[mp], adj[mp], feats, Fm);

        // nxt1 = relu(Eg@CeS0 + Fm@CeN0)      M=131072  (dominant GEMM)
        k_dual_gemm_wmma<DEDGE, DDIM><<<NEDGEROW / 64, 128, 0, stream>>>(
            Eg, CeS0, Fm, CeN0, nxt1, nullptr, 0, 0);

        // nxt0 = relu(F0@CnS0 + Em@CnN0)      M=4096; also layer-0 output slice
        k_dual_gemm_wmma<DDIM, DEDGE><<<BSEED / 64, 128, 0, stream>>>(
            F0, CnS0, Em, CnN0,
            nxt0, out + (size_t)mp * BSEED * 256, 256, 0);

        // m1 = mean over 32 of nxt1 (post-relu, must materialize)
        k_mean32<<<BSEED, 128, 0, stream>>>(nxt1, m1);

        // out2 = relu(nxt0@Wn_self1 + m1@Wn_neigh1) -> layer-1 output slice
        k_dual_gemm_wmma<DDIM, DDIM><<<BSEED / 64, 128, 0, stream>>>(
            nxt0, Wn1s, m1, Wn1n,
            nullptr, out + (size_t)mp * BSEED * 256, 256, 128);
    }
}
